// Model_12171937316940
// MI455X (gfx1250) — matrix operand order repaired and checked by reading, not compiled, not run on hardware
//
#include <hip/hip_runtime.h>
#include <math.h>

// ---------------------------------------------------------------------------
// MeshGraphNets step for MI455X (gfx1250): f16 WMMA GEMMs + fused LN/scatter.
// N=8192 nodes, E=49152 edges, L=128, 15 processor steps.
// Weights are pre-shuffled once per launch into a WMMA-native layout so both
// A and B fragments load as contiguous ds_load_b128 pairs.
// ---------------------------------------------------------------------------

#define N_NODES 8192
#define N_EDGES 49152
#define LAT 128
#define STEPS 15

typedef __attribute__((ext_vector_type(16))) _Float16 v16h;
typedef __attribute__((ext_vector_type(8)))  float    v8f;

// ------------------------- weight shuffling --------------------------------
// src: [steps, Krows, 128] f32 row-major; step = blockIdx.y.
// dst: [steps, Kpad/32, 128, 32] f16 where dst[s][kc][n][ko] = src[s][kc*32+ko][n]
// (zero for kc*32+ko >= Krows). Makes the WMMA B fragment lane-contiguous.
__global__ void k_conv_wmma_layout(const float* __restrict__ src, _Float16* __restrict__ dst,
                                   int Krows, int Kpad) {
  const int s = blockIdx.y;
  const int per = Kpad << 7;                       // Kpad*128
  const int r = blockIdx.x * blockDim.x + threadIdx.x;
  if (r >= per) return;
  const int kc  = r >> 12;                         // / 4096
  const int rem = r & 4095;
  const int n   = rem >> 5;
  const int ko  = rem & 31;
  const int k   = kc * 32 + ko;
  dst[(long)s * per + r] =
      (k < Krows) ? (_Float16)src[((long)s * Krows + k) * 128 + n] : (_Float16)0.0f;
}

__global__ void k_zero_f32(float* __restrict__ p, int n) {
  int i = blockIdx.x * blockDim.x + threadIdx.x;
  if (i < n) p[i] = 0.0f;
}

// ------------------------- graph feature builders --------------------------

__global__ void k_build_node_feat(const float* __restrict__ wp, const float* __restrict__ pwp,
                                  const int* __restrict__ nt,
                                  const float* __restrict__ mean, const float* __restrict__ stdv,
                                  _Float16* __restrict__ nf, int n) {
  int i = blockIdx.x * blockDim.x + threadIdx.x;
  if (i >= n) return;
  float f[12];
  f[0] = wp[i * 3 + 0] - pwp[i * 3 + 0];
  f[1] = wp[i * 3 + 1] - pwp[i * 3 + 1];
  f[2] = wp[i * 3 + 2] - pwp[i * 3 + 2];
  int t = nt[i];
#pragma unroll
  for (int k = 0; k < 9; ++k) f[3 + k] = (t == k) ? 1.0f : 0.0f;
#pragma unroll
  for (int k = 0; k < 12; ++k)
    nf[(long)i * 32 + k] = (_Float16)((f[k] - mean[k]) / stdv[k]);
#pragma unroll
  for (int k = 12; k < 32; ++k) nf[(long)i * 32 + k] = (_Float16)0.0f;
}

__global__ void k_build_edge_feat(const float* __restrict__ wp, const float* __restrict__ mp,
                                  const int* __restrict__ snd, const int* __restrict__ rcv,
                                  const float* __restrict__ mean, const float* __restrict__ stdv,
                                  _Float16* __restrict__ ef, int n) {
  int e = blockIdx.x * blockDim.x + threadIdx.x;
  if (e >= n) return;
  int s = snd[e], r = rcv[e];
  float rw0 = wp[s * 3 + 0] - wp[r * 3 + 0];
  float rw1 = wp[s * 3 + 1] - wp[r * 3 + 1];
  float rw2 = wp[s * 3 + 2] - wp[r * 3 + 2];
  float rm0 = mp[s * 3 + 0] - mp[r * 3 + 0];
  float rm1 = mp[s * 3 + 1] - mp[r * 3 + 1];
  float rm2 = mp[s * 3 + 2] - mp[r * 3 + 2];
  float f[8];
  f[0] = rw0; f[1] = rw1; f[2] = rw2;
  f[3] = sqrtf(rw0 * rw0 + rw1 * rw1 + rw2 * rw2);
  f[4] = rm0; f[5] = rm1; f[6] = rm2;
  f[7] = sqrtf(rm0 * rm0 + rm1 * rm1 + rm2 * rm2);
#pragma unroll
  for (int k = 0; k < 8; ++k)
    ef[(long)e * 32 + k] = (_Float16)((f[k] - mean[k]) / stdv[k]);
#pragma unroll
  for (int k = 8; k < 32; ++k) ef[(long)e * 32 + k] = (_Float16)0.0f;
}

// edge MLP input: concat(e_lat, n_lat[snd], n_lat[rcv]) -> [E,384] f16.
// One 16B chunk (8 halves) per thread; 48 chunks per edge. 32-bit indexing.
__global__ void k_build_edge_input(const _Float16* __restrict__ e16, const _Float16* __restrict__ n16,
                                   const int* __restrict__ snd, const int* __restrict__ rcv,
                                   _Float16* __restrict__ ein, int nChunks) {
  int i = blockIdx.x * blockDim.x + threadIdx.x;
  if (i >= nChunks) return;
  int e = i / 48;                    // constant division -> mul/shift
  int c = (i - e * 48) * 8;
  const _Float16* sp;
  if (c < 128)      sp = e16 + (long)e * 128 + c;
  else if (c < 256) sp = n16 + (long)snd[e] * 128 + (c - 128);
  else              sp = n16 + (long)rcv[e] * 128 + (c - 256);
  *(uint4*)(ein + (long)e * 384 + c) = *(const uint4*)sp;
}

// node MLP input: concat(n_lat, agg) -> [N,256] f16. 32 chunks of 8 per node.
__global__ void k_build_node_input(const _Float16* __restrict__ n16, const float* __restrict__ agg,
                                   _Float16* __restrict__ nin, int nChunks) {
  int i = blockIdx.x * blockDim.x + threadIdx.x;
  if (i >= nChunks) return;
  int n = i >> 5;
  int c = (i & 31) * 8;
  if (c < 128) {
    *(uint4*)(nin + (long)n * 256 + c) = *(const uint4*)(n16 + (long)n * 128 + c);
  } else {
    const float* a = agg + (long)n * 128 + (c - 128);
    _Float16* d = nin + (long)n * 256 + c;
#pragma unroll
    for (int t = 0; t < 8; ++t) d[t] = (_Float16)a[t];
  }
}

// ------------------------- WMMA GEMM (the workhorse) -----------------------
// C[M,128] = A[M,K] @ W[K,128] + bias ; optional ReLU.
// A: [M,K] f16 row-major. Wt: WMMA-native layout [K/32][128][32] f16.
// Block: 256 threads (8 wave32), tile 128x128, double-buffered LDS.
__global__ __launch_bounds__(256)
void gemm128(const _Float16* __restrict__ Aacts, const _Float16* __restrict__ Wt,
             const float* __restrict__ bias, int K,
             _Float16* __restrict__ outH, float* __restrict__ outF, int relu) {
  __shared__ __align__(16) _Float16 As[2][128 * 32];  // 2 x 8 KB
  __shared__ __align__(16) _Float16 Bs[2][128 * 32];  // 2 x 8 KB

  const int tid  = threadIdx.x;
  const int wave = tid >> 5;       // 0..7
  const int lane = tid & 31;
  const int n0   = lane & 15;
  const int hb   = lane >> 4;
  const long rowBlock = (long)blockIdx.x * 128;

  v8f acc[8] = {};

  const int nK = K >> 5;
  const int sr = tid >> 1;         // staging row (0..127)
  const int sc = (tid & 1) * 16;   // staging col chunk (halves)

  const _Float16* aSrc0 = Aacts + (rowBlock + sr) * (long)K + sc;
  const _Float16* bSrc0 = Wt + (long)sr * 32 + sc;
  _Float16* aDst[2] = { &As[0][sr * 32 + sc], &As[1][sr * 32 + sc] };
  _Float16* bDst[2] = { &Bs[0][sr * 32 + sc], &Bs[1][sr * 32 + sc] };

  // prologue: stage chunk 0 into buffer 0
  uint4 ra0 = ((const uint4*)aSrc0)[0], ra1 = ((const uint4*)aSrc0)[1];
  uint4 rb0 = ((const uint4*)bSrc0)[0], rb1 = ((const uint4*)bSrc0)[1];
  ((uint4*)aDst[0])[0] = ra0; ((uint4*)aDst[0])[1] = ra1;
  ((uint4*)bDst[0])[0] = rb0; ((uint4*)bDst[0])[1] = rb1;
  __syncthreads();

  const int am = wave * 16 + n0;

  for (int kc = 0; kc < nK; ++kc) {
    const int cur = kc & 1;
    const bool hasNext = (kc + 1) < nK;
    if (hasNext) {
      const _Float16* aS = aSrc0 + (long)(kc + 1) * 32;
      const _Float16* bS = bSrc0 + (long)(kc + 1) * 4096;
      ra0 = ((const uint4*)aS)[0]; ra1 = ((const uint4*)aS)[1];
      rb0 = ((const uint4*)bS)[0]; rb1 = ((const uint4*)bS)[1];
    }

    // A fragment (16-bit A 16x32): 16 contiguous halves at hb*16
    union { v16h v; _Float16 h[16]; unsigned int u[8]; } af;
    {
      const _Float16* ap = &As[cur][am * 32];
#pragma unroll
      for (int v = 0; v < 8; ++v) {
        const int kb = hb * 16 + v * 2;
        af.u[v] = *(const unsigned int*)(ap + kb);
      }
    }

#pragma unroll
    for (int j = 0; j < 8; ++j) {
      // B fragment: 16 contiguous halves at Bs[(j*16+n0)*32 + hb*16]
      union { v16h v; _Float16 h[16]; unsigned int u[8]; } bf;
      const _Float16* bp = &Bs[cur][(j * 16 + n0) * 32 + hb * 16];
#pragma unroll
      for (int v = 0; v < 8; ++v) bf.u[v] = *(const unsigned int*)(bp + 2 * v);
      acc[j] = __builtin_amdgcn_wmma_f32_16x16x32_f16(
          false, af.v, false, bf.v, (short)0, acc[j], false, false);
    }

    if (hasNext) {
      __syncthreads();  // all waves done reading buffer cur^1 (last read in kc-1)
      const int nxt = cur ^ 1;
      ((uint4*)aDst[nxt])[0] = ra0; ((uint4*)aDst[nxt])[1] = ra1;
      ((uint4*)bDst[nxt])[0] = rb0; ((uint4*)bDst[nxt])[1] = rb1;
      __syncthreads();
    }
  }

  // Epilogue: C/D layout — VGPR r holds (M = hb*8 + r, N = lane&15).
  const long rowBase = rowBlock + wave * 16 + hb * 8;
#pragma unroll
  for (int j = 0; j < 8; ++j) {
    const int col = j * 16 + n0;
    const float bv = bias ? bias[col] : 0.0f;
#pragma unroll
    for (int r = 0; r < 8; ++r) {
      float cv = acc[j][r] + bv;
      if (relu) cv = cv > 0.0f ? cv : 0.0f;
      const long oi = (rowBase + r) * 128 + col;
      if (outH) outH[oi] = (_Float16)cv;
      if (outF) outF[oi] = cv;
    }
  }
}

// --------------- fused LayerNorm (+ residual, + segment_sum scatter) -------
__global__ void k_layernorm(const float* __restrict__ x, const float* __restrict__ gamma,
                            const float* __restrict__ beta, float* __restrict__ lat,
                            _Float16* __restrict__ lat16, int residual,
                            const int* __restrict__ recv, float* __restrict__ agg) {
  __shared__ float red[128];
  const long row = blockIdx.x;
  const int c = threadIdx.x;
  const long idx = row * 128 + c;

  float v = x[idx];
  red[c] = v;
  __syncthreads();
#pragma unroll
  for (int s2 = 64; s2 > 0; s2 >>= 1) {
    if (c < s2) red[c] += red[c + s2];
    __syncthreads();
  }
  float mean = red[0] * (1.0f / 128.0f);
  __syncthreads();
  float d = v - mean;
  red[c] = d * d;
  __syncthreads();
#pragma unroll
  for (int s2 = 64; s2 > 0; s2 >>= 1) {
    if (c < s2) red[c] += red[c + s2];
    __syncthreads();
  }
  float var = red[0] * (1.0f / 128.0f);
  float u = d * rsqrtf(var + 1e-5f) * gamma[c] + beta[c];

  if (agg) atomicAdd(agg + (long)recv[row] * 128 + c, u);

  float nl = residual ? (lat[idx] + u) : u;
  lat[idx] = nl;
  lat16[idx] = (_Float16)nl;
}

// ------------------------- decoder final layer (128 -> 3) ------------------
__global__ void k_decoder_out(const _Float16* __restrict__ h, const float* __restrict__ W3,
                              const float* __restrict__ b3, float* __restrict__ out, int n) {
  int i = blockIdx.x * blockDim.x + threadIdx.x;
  if (i >= n) return;
  float a0 = b3[0], a1 = b3[1], a2 = b3[2];
  const _Float16* hr = h + (long)i * 128;
#pragma unroll 4
  for (int k = 0; k < 128; ++k) {
    float hv = (float)hr[k];
    a0 += hv * W3[k * 3 + 0];
    a1 += hv * W3[k * 3 + 1];
    a2 += hv * W3[k * 3 + 2];
  }
  out[i * 3 + 0] = a0;
  out[i * 3 + 1] = a1;
  out[i * 3 + 2] = a2;
}

// ---------------------------------------------------------------------------

extern "C" void kernel_launch(void* const* d_in, const int* in_sizes, int n_in,
                              void* d_out, int out_size, void* d_ws, size_t ws_size,
                              hipStream_t stream) {
  (void)in_sizes; (void)n_in; (void)out_size; (void)ws_size;

  // Inputs in setup_inputs() dict (insertion) order, recursing into params/proc.
  const float* world_pos      = (const float*)d_in[0];
  const float* prev_world_pos = (const float*)d_in[1];
  const float* mesh_pos       = (const float*)d_in[2];
  const float* node_mean = (const float*)d_in[3];
  const float* node_std  = (const float*)d_in[4];
  const float* edge_mean = (const float*)d_in[5];
  const float* edge_std  = (const float*)d_in[6];
  const float* en_W1 = (const float*)d_in[7];
  const float* en_b1 = (const float*)d_in[8];
  const float* en_W2 = (const float*)d_in[9];
  const float* en_b2 = (const float*)d_in[10];
  const float* en_W3 = (const float*)d_in[11];
  const float* en_b3 = (const float*)d_in[12];
  const float* en_g  = (const float*)d_in[13];
  const float* en_bt = (const float*)d_in[14];
  const float* ee_W1 = (const float*)d_in[15];
  const float* ee_b1 = (const float*)d_in[16];
  const float* ee_W2 = (const float*)d_in[17];
  const float* ee_b2 = (const float*)d_in[18];
  const float* ee_W3 = (const float*)d_in[19];
  const float* ee_b3 = (const float*)d_in[20];
  const float* ee_g  = (const float*)d_in[21];
  const float* ee_bt = (const float*)d_in[22];
  const float* p_eW1 = (const float*)d_in[23];
  const float* p_eb1 = (const float*)d_in[24];
  const float* p_eW2 = (const float*)d_in[25];
  const float* p_eb2 = (const float*)d_in[26];
  const float* p_eW3 = (const float*)d_in[27];
  const float* p_eb3 = (const float*)d_in[28];
  const float* p_eg  = (const float*)d_in[29];
  const float* p_ebt = (const float*)d_in[30];
  const float* p_nW1 = (const float*)d_in[31];
  const float* p_nb1 = (const float*)d_in[32];
  const float* p_nW2 = (const float*)d_in[33];
  const float* p_nb2 = (const float*)d_in[34];
  const float* p_nW3 = (const float*)d_in[35];
  const float* p_nb3 = (const float*)d_in[36];
  const float* p_ng  = (const float*)d_in[37];
  const float* p_nbt = (const float*)d_in[38];
  const float* de_W1 = (const float*)d_in[39];
  const float* de_b1 = (const float*)d_in[40];
  const float* de_W2 = (const float*)d_in[41];
  const float* de_b2 = (const float*)d_in[42];
  const float* de_W3 = (const float*)d_in[43];
  const float* de_b3 = (const float*)d_in[44];
  const int* node_type = (const int*)d_in[45];
  const int* senders   = (const int*)d_in[46];
  const int* receivers = (const int*)d_in[47];
  float* out = (float*)d_out;

  // ---- workspace carve-out (256B-aligned) ----
  char* base = (char*)d_ws;
  size_t off = 0;
  auto carve = [&](size_t bytes) -> void* {
    void* p = base + off;
    off = (off + bytes + 255) & ~((size_t)255);
    return p;
  };
  _Float16* w_enW1 = (_Float16*)carve((size_t)32 * 128 * 2);
  _Float16* w_enW2 = (_Float16*)carve((size_t)128 * 128 * 2);
  _Float16* w_enW3 = (_Float16*)carve((size_t)128 * 128 * 2);
  _Float16* w_eeW1 = (_Float16*)carve((size_t)32 * 128 * 2);
  _Float16* w_eeW2 = (_Float16*)carve((size_t)128 * 128 * 2);
  _Float16* w_eeW3 = (_Float16*)carve((size_t)128 * 128 * 2);
  _Float16* w_eW1  = (_Float16*)carve((size_t)STEPS * 384 * 128 * 2);
  _Float16* w_eW2  = (_Float16*)carve((size_t)STEPS * 128 * 128 * 2);
  _Float16* w_eW3  = (_Float16*)carve((size_t)STEPS * 128 * 128 * 2);
  _Float16* w_nW1  = (_Float16*)carve((size_t)STEPS * 256 * 128 * 2);
  _Float16* w_nW2  = (_Float16*)carve((size_t)STEPS * 128 * 128 * 2);
  _Float16* w_nW3  = (_Float16*)carve((size_t)STEPS * 128 * 128 * 2);
  _Float16* w_dW1  = (_Float16*)carve((size_t)128 * 128 * 2);
  _Float16* w_dW2  = (_Float16*)carve((size_t)128 * 128 * 2);
  _Float16* nf16 = (_Float16*)carve((size_t)N_NODES * 32 * 2);
  _Float16* ef16 = (_Float16*)carve((size_t)N_EDGES * 32 * 2);
  _Float16* n16  = (_Float16*)carve((size_t)N_NODES * 128 * 2);
  float*    nlat = (float*)carve((size_t)N_NODES * 128 * 4);
  _Float16* e16  = (_Float16*)carve((size_t)N_EDGES * 128 * 2);
  float*    elat = (float*)carve((size_t)N_EDGES * 128 * 4);
  _Float16* ein16 = (_Float16*)carve((size_t)N_EDGES * 384 * 2);
  _Float16* t1   = (_Float16*)carve((size_t)N_EDGES * 128 * 2);
  _Float16* t2   = (_Float16*)carve((size_t)N_EDGES * 128 * 2);
  float*    t3f  = (float*)carve((size_t)N_EDGES * 128 * 4);
  float*    agg  = (float*)carve((size_t)N_NODES * 128 * 4);
  _Float16* nin16 = (_Float16*)carve((size_t)N_NODES * 256 * 2);
  _Float16* u1   = (_Float16*)carve((size_t)N_NODES * 128 * 2);
  _Float16* u2   = (_Float16*)carve((size_t)N_NODES * 128 * 2);
  float*    u3f  = (float*)carve((size_t)N_NODES * 128 * 4);

  dim3 b256(256);
#define GRID1(n) dim3((unsigned int)((((long)(n)) + 255) / 256))
#define GRIDW(kpad, steps) dim3((unsigned int)((((kpad) << 7) + 255) / 256), (unsigned int)(steps))

  // ---- weights f32 -> WMMA-native f16 layout (K padded to mult of 32) ----
  k_conv_wmma_layout<<<GRIDW(32, 1), b256, 0, stream>>>(en_W1, w_enW1, 12, 32);
  k_conv_wmma_layout<<<GRIDW(32, 1), b256, 0, stream>>>(ee_W1, w_eeW1, 8, 32);
  k_conv_wmma_layout<<<GRIDW(128, 1), b256, 0, stream>>>(en_W2, w_enW2, 128, 128);
  k_conv_wmma_layout<<<GRIDW(128, 1), b256, 0, stream>>>(en_W3, w_enW3, 128, 128);
  k_conv_wmma_layout<<<GRIDW(128, 1), b256, 0, stream>>>(ee_W2, w_eeW2, 128, 128);
  k_conv_wmma_layout<<<GRIDW(128, 1), b256, 0, stream>>>(ee_W3, w_eeW3, 128, 128);
  k_conv_wmma_layout<<<GRIDW(384, STEPS), b256, 0, stream>>>(p_eW1, w_eW1, 384, 384);
  k_conv_wmma_layout<<<GRIDW(128, STEPS), b256, 0, stream>>>(p_eW2, w_eW2, 128, 128);
  k_conv_wmma_layout<<<GRIDW(128, STEPS), b256, 0, stream>>>(p_eW3, w_eW3, 128, 128);
  k_conv_wmma_layout<<<GRIDW(256, STEPS), b256, 0, stream>>>(p_nW1, w_nW1, 256, 256);
  k_conv_wmma_layout<<<GRIDW(128, STEPS), b256, 0, stream>>>(p_nW2, w_nW2, 128, 128);
  k_conv_wmma_layout<<<GRIDW(128, STEPS), b256, 0, stream>>>(p_nW3, w_nW3, 128, 128);
  k_conv_wmma_layout<<<GRIDW(128, 1), b256, 0, stream>>>(de_W1, w_dW1, 128, 128);
  k_conv_wmma_layout<<<GRIDW(128, 1), b256, 0, stream>>>(de_W2, w_dW2, 128, 128);

  // ---- build graph features ----
  k_build_node_feat<<<GRID1(N_NODES), b256, 0, stream>>>(world_pos, prev_world_pos, node_type,
                                                         node_mean, node_std, nf16, N_NODES);
  k_build_edge_feat<<<GRID1(N_EDGES), b256, 0, stream>>>(world_pos, mesh_pos, senders, receivers,
                                                         edge_mean, edge_std, ef16, N_EDGES);

  const dim3 gN(N_NODES / 128), gE(N_EDGES / 128);

  // ---- encoders ----
  gemm128<<<gN, b256, 0, stream>>>(nf16, w_enW1, en_b1, 32, u1, nullptr, 1);
  gemm128<<<gN, b256, 0, stream>>>(u1, w_enW2, en_b2, 128, u2, nullptr, 1);
  gemm128<<<gN, b256, 0, stream>>>(u2, w_enW3, en_b3, 128, nullptr, u3f, 0);
  k_layernorm<<<dim3(N_NODES), dim3(128), 0, stream>>>(u3f, en_g, en_bt, nlat, n16, 0, nullptr, nullptr);

  gemm128<<<gE, b256, 0, stream>>>(ef16, w_eeW1, ee_b1, 32, t1, nullptr, 1);
  gemm128<<<gE, b256, 0, stream>>>(t1, w_eeW2, ee_b2, 128, t2, nullptr, 1);
  gemm128<<<gE, b256, 0, stream>>>(t2, w_eeW3, ee_b3, 128, nullptr, t3f, 0);
  k_layernorm<<<dim3(N_EDGES), dim3(128), 0, stream>>>(t3f, ee_g, ee_bt, elat, e16, 0, nullptr, nullptr);

  // ---- processor: 15 GraphNetBlocks ----
  for (int s = 0; s < STEPS; ++s) {
    const _Float16* eW1 = w_eW1 + (size_t)s * 384 * 128;
    const _Float16* eW2 = w_eW2 + (size_t)s * 128 * 128;
    const _Float16* eW3 = w_eW3 + (size_t)s * 128 * 128;
    const _Float16* nW1 = w_nW1 + (size_t)s * 256 * 128;
    const _Float16* nW2 = w_nW2 + (size_t)s * 128 * 128;
    const _Float16* nW3 = w_nW3 + (size_t)s * 128 * 128;

    k_build_edge_input<<<GRID1(N_EDGES * 48), b256, 0, stream>>>(
        e16, n16, senders, receivers, ein16, N_EDGES * 48);
    gemm128<<<gE, b256, 0, stream>>>(ein16, eW1, p_eb1 + s * 128, 384, t1, nullptr, 1);
    gemm128<<<gE, b256, 0, stream>>>(t1, eW2, p_eb2 + s * 128, 128, t2, nullptr, 1);
    gemm128<<<gE, b256, 0, stream>>>(t2, eW3, p_eb3 + s * 128, 128, nullptr, t3f, 0);

    k_zero_f32<<<GRID1(N_NODES * 128), b256, 0, stream>>>(agg, N_NODES * 128);
    // LN + residual on edge latent, fused with segment_sum(e_upd) scatter
    k_layernorm<<<dim3(N_EDGES), dim3(128), 0, stream>>>(
        t3f, p_eg + s * 128, p_ebt + s * 128, elat, e16, 1, receivers, agg);

    k_build_node_input<<<GRID1(N_NODES * 32), b256, 0, stream>>>(
        n16, agg, nin16, N_NODES * 32);
    gemm128<<<gN, b256, 0, stream>>>(nin16, nW1, p_nb1 + s * 128, 256, u1, nullptr, 1);
    gemm128<<<gN, b256, 0, stream>>>(u1, nW2, p_nb2 + s * 128, 128, u2, nullptr, 1);
    gemm128<<<gN, b256, 0, stream>>>(u2, nW3, p_nb3 + s * 128, 128, nullptr, u3f, 0);
    k_layernorm<<<dim3(N_NODES), dim3(128), 0, stream>>>(
        u3f, p_ng + s * 128, p_nbt + s * 128, nlat, n16, 1, nullptr, nullptr);
  }

  // ---- decoder ----
  gemm128<<<gN, b256, 0, stream>>>(n16, w_dW1, de_b1, 128, u1, nullptr, 1);
  gemm128<<<gN, b256, 0, stream>>>(u1, w_dW2, de_b2, 128, u2, nullptr, 1);
  k_decoder_out<<<GRID1(N_NODES), b256, 0, stream>>>(u2, de_W3, de_b3, out, N_NODES);
#undef GRID1
#undef GRIDW
}
